// InformerTimeMarching_35485019799715
// MI455X (gfx1250) — compile-verified, matrix operand order checked by reading, not run on hardware
//
#include <hip/hip_runtime.h>
#include <math.h>

// ---------------------------------------------------------------------------
// Informer forward pass for MI455X (gfx1250, wave32, WMMA).
// Dense GEMMs (QKV/O projections, FFN, K/V cross projections) run on
// V_WMMA_F32_16X16X32_F16 (f32 accumulate) with async global->LDS staging
// (ASYNCcnt) and double-buffered LDS tiles. Everything else (ProbSparse
// top-k, softmax, LayerNorm, distil conv, embedding) is wave32 VALU with
// LDS / lane-shuffle reductions.
// ---------------------------------------------------------------------------

typedef __attribute__((ext_vector_type(16))) _Float16 v16h;
typedef __attribute__((ext_vector_type(8)))  float    v8f;

#define DM   128      // d_model
#define NH   4        // heads
#define DH   32       // head dim (== wave32 lanes, one lane per dim)
#define BSZ  64       // batch

// gfx1250 async global->LDS copy, 16B per lane (tracked by ASYNCcnt).
// lds_off: LDS byte offset (dynamic-LDS base assumed 0: no static LDS here).
__device__ __forceinline__ void async_copy_b128(unsigned lds_off,
                                                const float* gptr) {
  asm volatile("global_load_async_to_lds_b128 %0, %1, off"
               :: "v"(lds_off), "v"((unsigned long long)(__SIZE_TYPE__)gptr)
               : "memory");
}

// ---------------------------------------------------------------------------
// WMMA GEMM: C[M,N] = act(A[M,K] @ W[K,N] + bias).
// Block = 128 threads (4 waves) -> 64x64 C tile; wave w owns rows
// [w*16, w*16+16) x 64 cols = 4 accumulators -> 4 WMMAs per K-step.
// A/B tiles staged f32 in LDS by async copies, double buffered.
// Requires M%64==0, N%64==0, K%32==0 (true for every call site here).
// act: 0 = none, 1 = exact GELU.
// ---------------------------------------------------------------------------
__global__ void gemm_wmma_kernel(const float* __restrict__ A,
                                 const float* __restrict__ W,
                                 const float* __restrict__ bias,
                                 float* __restrict__ C,
                                 int M, int N, int K, int act) {
  extern __shared__ float smem[];        // 2 x (A: 64x32 f32 | B: 32x64 f32)
  const int tid  = threadIdx.x;          // 0..127
  const int wv   = tid >> 5;             // wave 0..3
  const int lane = tid & 31;
  const int r    = lane & 15;            // row (A) / col (B) within 16-tile
  const int kh   = lane >> 4;            // lane-half selects K chunk
  const int m0   = blockIdx.y * 64;
  const int n0   = blockIdx.x * 64;

  // Cooperatively issue async copies for one K-step tile into buffer ib.
  auto issue = [&](int k0, int ib) {
    const unsigned baseOff = (unsigned)ib * 16384u;
#pragma unroll
    for (int it = 0; it < 4; ++it) {     // A: 64 rows x 32 K, f32
      int f = (it * 128 + tid) * 4;      // flat f32 index, 16B granules
      int row = f >> 5, kk = f & 31;
      async_copy_b128(baseOff + (unsigned)f * 4u,
                      A + (size_t)(m0 + row) * K + k0 + kk);
    }
#pragma unroll
    for (int it = 0; it < 4; ++it) {     // B: 32 K x 64 cols, f32
      int f = (it * 128 + tid) * 4;
      int kk = f >> 6, c = f & 63;
      async_copy_b128(baseOff + 8192u + (unsigned)f * 4u,
                      W + (size_t)(k0 + kk) * N + n0 + c);
    }
  };

  v8f acc[4] = {};
  const int nk = K / 32;
  issue(0, 0);
  for (int i = 0; i < nk; ++i) {
    const int ib = i & 1;
    if (i + 1 < nk) {
      issue((i + 1) * 32, ib ^ 1);
      asm volatile("s_wait_asynccnt 0x8" ::: "memory");  // drain tile i only
    } else {
      asm volatile("s_wait_asynccnt 0x0" ::: "memory");
    }
    __syncthreads();                     // all waves' copies of tile i done

    const float* sA = smem + ib * 4096;  // 16384 B
    const float* sB = sA + 2048;

    // A fragment (16x32 f16 layout, §7.12.2): lanes 0-15 K=j/16+j, +8 for hi.
    v16h a;
    const float* arow = sA + (wv * 16 + r) * 32 + kh * 8;
#pragma unroll
    for (int j = 0; j < 8; ++j) {
      a[j]     = (_Float16)arow[j];
      a[j + 8] = (_Float16)arow[16 + j];
    }
#pragma unroll
    for (int t = 0; t < 4; ++t) {        // 4 N-tiles reuse the A fragment
      v16h b;
      const float* bcol = sB + kh * 8 * 64 + t * 16 + r;
#pragma unroll
      for (int j = 0; j < 8; ++j) {
        b[j]     = (_Float16)bcol[j * 64];
        b[j + 8] = (_Float16)bcol[(16 + j) * 64];
      }
      acc[t] = __builtin_amdgcn_wmma_f32_16x16x32_f16(
          false, a, false, b, (short)0, acc[t], false, false);
    }
    __syncthreads();                     // reads done before buffer reuse
  }

  // C/D layout: VGPR v -> row v (lanes 0-15) / v+8 (lanes 16-31), col = lane&15
#pragma unroll
  for (int t = 0; t < 4; ++t) {
    const int col  = n0 + t * 16 + r;
    const float bv = bias ? bias[col] : 0.f;
#pragma unroll
    for (int vv = 0; vv < 8; ++vv) {
      const int row = m0 + wv * 16 + kh * 8 + vv;
      float val = acc[t][vv] + bv;
      if (act == 1) val = 0.5f * val * (1.0f + erff(val * 0.70710678118654752f));
      C[(size_t)row * N + col] = val;
    }
  }
}

// ---------------------------------------------------------------------------
// Data embedding: circular conv1d over time (k=3) + sinusoidal PE + TEMP_PE0.
// ---------------------------------------------------------------------------
__global__ void embed_kernel(const float* __restrict__ x,
                             const float* __restrict__ W,   // [DM, Cin, 3]
                             float* __restrict__ out,
                             int B, int L, int Cin) {
  int i = blockIdx.x * blockDim.x + threadIdx.x;
  if (i >= B * L * DM) return;
  int d = i % DM, l = (i / DM) % L, b = i / (DM * L);
  float acc = 0.f;
  for (int t = 0; t < 3; ++t) {
    int ls = (l - 1 + t + L) % L;
    const float* xr = x + (size_t)(b * L + ls) * Cin;
    const float* wr = W + (size_t)d * Cin * 3 + t;
    for (int c = 0; c < Cin; ++c) acc += xr[c] * wr[c * 3];
  }
  int pair  = d >> 1;
  float div = expf((float)(2 * pair) * (-9.210340371976184f / 128.f)); // -ln(1e4)/d
  float ang = (float)l * div;
  float pe  = (d & 1) ? cosf(ang) : sinf(ang);
  float tpe = (d & 1) ? 5.0f : 0.0f;   // 5 * sinusoid(pos=0): sin(0)=0, cos(0)=1
  out[i] = acc + pe + tpe;
}

// ---------------------------------------------------------------------------
// Fused residual add + LayerNorm. One wave per 128-wide row, 4 elems/lane.
// ---------------------------------------------------------------------------
__global__ void add_ln_kernel(const float* __restrict__ x,
                              const float* __restrict__ r,   // may be null
                              const float* __restrict__ g,
                              const float* __restrict__ bta,
                              float* __restrict__ out, int rows) {
  int wave = threadIdx.x >> 5, lane = threadIdx.x & 31;
  int row  = blockIdx.x * 8 + wave;
  if (row >= rows) return;
  const float* xr = x + (size_t)row * DM;
  float v[4];
#pragma unroll
  for (int j = 0; j < 4; ++j) {
    int d = lane * 4 + j;
    float t = xr[d];
    if (r) t += r[(size_t)row * DM + d];
    v[j] = t;
  }
  float s1 = v[0] + v[1] + v[2] + v[3];
  float s2 = v[0]*v[0] + v[1]*v[1] + v[2]*v[2] + v[3]*v[3];
  for (int m = 16; m > 0; m >>= 1) {
    s1 += __shfl_xor(s1, m, 32);
    s2 += __shfl_xor(s2, m, 32);
  }
  float mean = s1 * (1.f / 128.f);
  float var  = s2 * (1.f / 128.f) - mean * mean;
  float inv  = rsqrtf(var + 1e-5f);
#pragma unroll
  for (int j = 0; j < 4; ++j) {
    int d = lane * 4 + j;
    out[(size_t)row * DM + d] = (v[j] - mean) * inv * g[d] + bta[d];
  }
}

// ---------------------------------------------------------------------------
// ProbSparse attention pieces. Sampling uses a deterministic hash in place of
// JAX threefry (no runtime validation available); the sparse structure and
// update-set semantics of the reference are preserved.
// ---------------------------------------------------------------------------
__device__ __forceinline__ unsigned hash3(unsigned a, unsigned b, unsigned c) {
  unsigned h = a * 1664525u ^ b * 1013904223u ^ c * 747796405u;
  h ^= h >> 13; h *= 0x5bd1e995u; h ^= h >> 15;
  return h;
}

// M[b,h,lq] = max_s(q.k_sample) - sum_s(q.k_sample)/L_K
__global__ void prob_m_kernel(const float* __restrict__ q,
                              const float* __restrict__ k,
                              float* __restrict__ Mout,
                              int B, int LQ, int LK, int Up, unsigned seed) {
  int i = blockIdx.x * blockDim.x + threadIdx.x;
  if (i >= B * NH * LQ) return;
  int lq = i % LQ, h = (i / LQ) % NH, b = i / (LQ * NH);
  const float* qp = q + (size_t)(b * LQ + lq) * DM + h * DH;
  float mx = -3.4e38f, sm = 0.f;
  for (int s = 0; s < Up; ++s) {
    int kid = (int)(hash3((unsigned)lq, (unsigned)s, seed) % (unsigned)LK);
    const float* kp = k + (size_t)(b * LK + kid) * DM + h * DH;
    float d = 0.f;
    for (int t = 0; t < DH; ++t) d += qp[t] * kp[t];
    mx = fmaxf(mx, d); sm += d;
  }
  Mout[i] = mx - sm / (float)LK;
}

// Per-(b,h) iterative top-u selection (u <= 30, L <= 200) with bitmask.
__global__ void topk_kernel(const float* __restrict__ M, int* __restrict__ top,
                            int BH, int L, int u) {
  int i = blockIdx.x * blockDim.x + threadIdx.x;
  if (i >= BH) return;
  const float* m = M + (size_t)i * L;
  unsigned long long used[4] = {0ull, 0ull, 0ull, 0ull};
  for (int t = 0; t < u; ++t) {
    float best = -3.4e38f; int bi = 0;
    for (int l = 0; l < L; ++l) {
      if ((used[l >> 6] >> (l & 63)) & 1ull) continue;
      float v = m[l];
      if (v > best) { best = v; bi = l; }
    }
    used[bi >> 6] |= 1ull << (bi & 63);
    top[(size_t)i * u + t] = bi;
  }
}

// Encoder default context: broadcast mean_l v.
__global__ void mean_fill_kernel(const float* __restrict__ v,
                                 float* __restrict__ ctx, int B, int L) {
  int i = blockIdx.x * blockDim.x + threadIdx.x;
  if (i >= B * DM) return;
  int d = i % DM, b = i / DM;
  float s = 0.f;
  for (int l = 0; l < L; ++l) s += v[(size_t)(b * L + l) * DM + d];
  s /= (float)L;
  for (int l = 0; l < L; ++l) ctx[(size_t)(b * L + l) * DM + d] = s;
}

// Decoder (causal) default context: cumsum_l v.
__global__ void cumsum_kernel(const float* __restrict__ v,
                              float* __restrict__ ctx, int B, int L) {
  int i = blockIdx.x * blockDim.x + threadIdx.x;
  if (i >= B * DM) return;
  int d = i % DM, b = i / DM;
  float run = 0.f;
  for (int l = 0; l < L; ++l) {
    run += v[(size_t)(b * L + l) * DM + d];
    ctx[(size_t)(b * L + l) * DM + d] = run;
  }
}

// Softmax(q_top . K^T / sqrt(DH)) . V for each selected query. One wave per
// (b,h,ui); LDS stages the (<=200) scores; lane == head dim for the V pass.
__global__ void prob_update_kernel(const float* __restrict__ q,
                                   const float* __restrict__ k,
                                   const float* __restrict__ v,
                                   const int* __restrict__ top,
                                   float* __restrict__ ctx,
                                   int B, int LQ, int LK, int u, int causal) {
  __shared__ float sc[224];
  int blk = blockIdx.x;
  int ui = blk % u, h = (blk / u) % NH, b = blk / (u * NH);
  int lane = threadIdx.x;
  int lq = top[(size_t)(b * NH + h) * u + ui];
  int limit = causal ? (lq + 1 < LK ? lq + 1 : LK) : LK;
  const float* qp = q + (size_t)(b * LQ + lq) * DM + h * DH;

  float lmax = -3.4e38f;
  for (int j = lane; j < limit; j += 32) {
    const float* kp = k + (size_t)(b * LK + j) * DM + h * DH;
    float d = 0.f;
    for (int t = 0; t < DH; ++t) d += qp[t] * kp[t];
    d *= 0.17677669529663687f;   // 1/sqrt(32)
    sc[j] = d;
    lmax = fmaxf(lmax, d);
  }
  for (int m = 16; m > 0; m >>= 1) lmax = fmaxf(lmax, __shfl_xor(lmax, m, 32));

  float lsum = 0.f;
  for (int j = lane; j < limit; j += 32) {
    float e = expf(sc[j] - lmax);
    sc[j] = e;
    lsum += e;
  }
  for (int m = 16; m > 0; m >>= 1) lsum += __shfl_xor(lsum, m, 32);
  __syncthreads();   // all exp() scores visible before the V pass

  float acc = 0.f;
  for (int j = 0; j < limit; ++j)
    acc += sc[j] * v[(size_t)(b * LK + j) * DM + h * DH + lane];
  ctx[(size_t)(b * LQ + lq) * DM + h * DH + lane] = acc / lsum;
}

// Dense cross-attention: one wave per (b,h,lq), S <= 50 keys.
__global__ void full_attn_kernel(const float* __restrict__ q,
                                 const float* __restrict__ k,
                                 const float* __restrict__ v,
                                 float* __restrict__ ctx,
                                 int B, int LQ, int S) {
  __shared__ float sc[64];
  int blk = blockIdx.x;
  int lq = blk % LQ, h = (blk / LQ) % NH, b = blk / (LQ * NH);
  int lane = threadIdx.x;
  const float* qp = q + (size_t)(b * LQ + lq) * DM + h * DH;

  float lmax = -3.4e38f;
  for (int j = lane; j < S; j += 32) {
    const float* kp = k + (size_t)(b * S + j) * DM + h * DH;
    float d = 0.f;
    for (int t = 0; t < DH; ++t) d += qp[t] * kp[t];
    d *= 0.17677669529663687f;
    sc[j] = d;
    lmax = fmaxf(lmax, d);
  }
  for (int m = 16; m > 0; m >>= 1) lmax = fmaxf(lmax, __shfl_xor(lmax, m, 32));
  float lsum = 0.f;
  for (int j = lane; j < S; j += 32) {
    float e = expf(sc[j] - lmax);
    sc[j] = e;
    lsum += e;
  }
  for (int m = 16; m > 0; m >>= 1) lsum += __shfl_xor(lsum, m, 32);
  __syncthreads();
  float acc = 0.f;
  for (int j = 0; j < S; ++j)
    acc += sc[j] * v[(size_t)(b * S + j) * DM + h * DH + lane];
  ctx[(size_t)(b * LQ + lq) * DM + h * DH + lane] = acc / lsum;
}

// Decoder "mix" permute: reshape (B,H,L,D) -> (B,L,H*D) scramble.
__global__ void mix_kernel(const float* __restrict__ ctx,
                           float* __restrict__ out, int B, int L) {
  int i = blockIdx.x * blockDim.x + threadIdx.x;
  if (i >= B * L * DM) return;
  int j = i % DM, l = (i / DM) % L, b = i / (DM * L);
  int f = l * DM + j;                 // flat index within (H,L,D)
  int h = f / (L * DH);
  int rem = f % (L * DH);
  int ls = rem / DH, d = rem % DH;
  out[i] = ctx[(size_t)(b * L + ls) * DM + h * DH + d];
}

// Distil: circular conv1d (128->128, k=3) + BN + ELU.
__global__ void distil_conv_kernel(const float* __restrict__ x,
                                   const float* __restrict__ W,  // [DM, DM, 3]
                                   const float* __restrict__ bias,
                                   const float* __restrict__ bng,
                                   const float* __restrict__ bnb,
                                   const float* __restrict__ bnm,
                                   const float* __restrict__ bnv,
                                   float* __restrict__ out, int B, int L) {
  int i = blockIdx.x * blockDim.x + threadIdx.x;
  if (i >= B * L * DM) return;
  int d = i % DM, l = (i / DM) % L, b = i / (DM * L);
  float acc = bias[d];
  const float* wd = W + (size_t)d * DM * 3;
  for (int t = 0; t < 3; ++t) {
    int ls = (l - 1 + t + L) % L;
    const float* xr = x + (size_t)(b * L + ls) * DM;
    const float* wt = wd + t;
    for (int c = 0; c < DM; ++c) acc += xr[c] * wt[c * 3];
  }
  float scale = bng[d] * rsqrtf(bnv[d] + 1e-5f);
  float y = (acc - bnm[d]) * scale + bnb[d];
  out[i] = y > 0.f ? y : (expf(y) - 1.f);   // ELU
}

// Maxpool k=3 stride=2 pad=1 over time.
__global__ void maxpool_kernel(const float* __restrict__ x,
                               float* __restrict__ out, int B, int L, int L2) {
  int i = blockIdx.x * blockDim.x + threadIdx.x;
  if (i >= B * L2 * DM) return;
  int d = i % DM, lo = (i / DM) % L2, b = i / (DM * L2);
  float m = -3.4e38f;
  for (int t = -1; t <= 1; ++t) {
    int l = 2 * lo + t;
    if (l >= 0 && l < L) m = fmaxf(m, x[(size_t)(b * L + l) * DM + d]);
  }
  out[i] = m;
}

// Final projection 128 -> 12 (N too small for a 16-wide WMMA tile).
__global__ void proj_kernel(const float* __restrict__ y,
                            const float* __restrict__ W,   // [128,12]
                            const float* __restrict__ bias,
                            float* __restrict__ out, int B, int L) {
  int i = blockIdx.x * blockDim.x + threadIdx.x;
  if (i >= B * L * 12) return;
  int c = i % 12, l = (i / 12) % L, b = i / (12 * L);
  const float* yr = y + (size_t)(b * L + l) * DM;
  float acc = bias[c];
  for (int t = 0; t < DM; ++t) acc += yr[t] * W[t * 12 + c];
  out[i] = acc;
}

// ---------------------------------------------------------------------------
// Host-side orchestration.
// ---------------------------------------------------------------------------
namespace {
struct AttnP { const float *Wq,*bq,*Wk,*bk,*Wv,*bv,*Wo,*bo; };
struct EncL  { AttnP a; const float *W1,*b1,*W2,*b2,*l1g,*l1b,*l2g,*l2b; };
struct ConvP { const float *W,*b,*g,*bb,*m,*v; };
struct DecL  { AttnP sa, ca; const float *W1,*b1,*W2,*b2,
               *l1g,*l1b,*l2g,*l2b,*l3g,*l3b; };

inline int ufac(int L) {   // min(5*ceil(ln(L)), L)
  int c = (int)ceil(log((double)L) - 1e-12);
  int v = 5 * c;
  return v < L ? v : L;
}
inline int cdiv(int a, int b) { return (a + b - 1) / b; }
}  // namespace

extern "C" void kernel_launch(void* const* d_in, const int* in_sizes, int n_in,
                              void* d_out, int out_size, void* d_ws, size_t ws_size,
                              hipStream_t stream) {
  (void)in_sizes; (void)n_in; (void)out_size; (void)ws_size;
  const float** in = (const float**)d_in;
  int p = 0;
  auto nxt = [&]() { return in[p++]; };

  // ---- unpack params in setup_inputs() recursive insertion order ----
  const float* x_enc = nxt();
  const float* x_dec = nxt();
  const float* enc_emb_W = nxt();
  const float* dec_emb_W = nxt();
  auto rdAttn = [&]() {
    AttnP a;
    a.Wq = nxt(); a.bq = nxt(); a.Wk = nxt(); a.bk = nxt();
    a.Wv = nxt(); a.bv = nxt(); a.Wo = nxt(); a.bo = nxt();
    return a;
  };
  EncL enc[3];
  for (int i = 0; i < 3; ++i) {
    enc[i].a = rdAttn();
    enc[i].W1 = nxt(); enc[i].b1 = nxt(); enc[i].W2 = nxt(); enc[i].b2 = nxt();
    enc[i].l1g = nxt(); enc[i].l1b = nxt(); enc[i].l2g = nxt(); enc[i].l2b = nxt();
  }
  ConvP cv[2];
  for (int i = 0; i < 2; ++i) {
    cv[i].W = nxt(); cv[i].b = nxt(); cv[i].g = nxt();
    cv[i].bb = nxt(); cv[i].m = nxt(); cv[i].v = nxt();
  }
  const float* enc_ng = nxt(); const float* enc_nb = nxt();
  DecL dec[2];
  for (int i = 0; i < 2; ++i) {
    dec[i].sa = rdAttn(); dec[i].ca = rdAttn();
    dec[i].W1 = nxt(); dec[i].b1 = nxt(); dec[i].W2 = nxt(); dec[i].b2 = nxt();
    dec[i].l1g = nxt(); dec[i].l1b = nxt();
    dec[i].l2g = nxt(); dec[i].l2b = nxt();
    dec[i].l3g = nxt(); dec[i].l3b = nxt();
  }
  const float* dec_ng = nxt(); const float* dec_nb = nxt();
  const float* projW  = nxt(); const float* projb  = nxt();

  // ---- workspace arena ----
  char* base = (char*)d_ws;
  size_t off = 0;
  auto alloc = [&](size_t nfloat) -> float* {
    off = (off + 255) & ~(size_t)255;
    float* q = (float*)(base + off);
    off += nfloat * sizeof(float);
    return q;
  };
  const size_t SZ1 = (size_t)BSZ * 200 * DM;       // 1.6M floats
  float* xe     = alloc(SZ1);
  float* xb2    = alloc(SZ1);   // also reused as the decoder "mixed" buffer
  float* crossb = alloc(SZ1);
  float* y1     = alloc(SZ1);
  float* y2     = alloc(SZ1);
  float* qb     = alloc(SZ1);
  float* kb     = alloc(SZ1);
  float* vb     = alloc(SZ1);
  float* ctx    = alloc(SZ1);
  float* pout   = alloc(SZ1);
  float* hid    = alloc((size_t)BSZ * 200 * 512);
  float* Mbuf   = alloc((size_t)BSZ * NH * 200);
  int*   topb   = (int*)alloc((size_t)BSZ * NH * 32);

  auto gemm = [&](const float* A, const float* W, const float* bias, float* C,
                  int M, int N, int K, int act) {
    dim3 g(N / 64, M / 64);
    gemm_wmma_kernel<<<g, 128, 32768, stream>>>(A, W, bias, C, M, N, K, act);
  };
  auto add_ln = [&](const float* x, const float* r, const float* g,
                    const float* b, float* o, int rows) {
    add_ln_kernel<<<cdiv(rows, 8), 256, 0, stream>>>(x, r, g, b, o, rows);
  };
  auto prob_attn = [&](const float* q, const float* k, const float* v,
                       float* c, int L, int causal, unsigned seed) {
    int u = ufac(L);
    prob_m_kernel<<<cdiv(BSZ * NH * L, 256), 256, 0, stream>>>(
        q, k, Mbuf, BSZ, L, L, u, seed);
    topk_kernel<<<cdiv(BSZ * NH, 64), 64, 0, stream>>>(Mbuf, topb, BSZ * NH, L, u);
    if (causal)
      cumsum_kernel<<<cdiv(BSZ * DM, 256), 256, 0, stream>>>(v, c, BSZ, L);
    else
      mean_fill_kernel<<<cdiv(BSZ * DM, 256), 256, 0, stream>>>(v, c, BSZ, L);
    prob_update_kernel<<<BSZ * NH * u, 32, 0, stream>>>(
        q, k, v, topb, c, BSZ, L, L, u, causal);
  };

  // ================= Encoder =================
  int L = 200;
  embed_kernel<<<cdiv(BSZ * L * DM, 256), 256, 0, stream>>>(
      x_enc, enc_emb_W, xe, BSZ, L, 12);
  for (int i = 0; i < 3; ++i) {
    int M = BSZ * L;
    gemm(xe, enc[i].a.Wq, enc[i].a.bq, qb, M, DM, DM, 0);
    gemm(xe, enc[i].a.Wk, enc[i].a.bk, kb, M, DM, DM, 0);
    gemm(xe, enc[i].a.Wv, enc[i].a.bv, vb, M, DM, DM, 0);
    prob_attn(qb, kb, vb, ctx, L, /*causal=*/0, (unsigned)i);
    gemm(ctx, enc[i].a.Wo, enc[i].a.bo, pout, M, DM, DM, 0);
    add_ln(xe, pout, enc[i].l1g, enc[i].l1b, xb2, M);
    gemm(xb2, enc[i].W1, enc[i].b1, hid, M, 512, DM, /*gelu=*/1);
    gemm(hid, enc[i].W2, enc[i].b2, pout, M, DM, 512, 0);
    add_ln(xb2, pout, enc[i].l2g, enc[i].l2b, xe, M);
    if (i < 2) {  // distil conv
      distil_conv_kernel<<<cdiv(BSZ * L * DM, 256), 256, 0, stream>>>(
          xe, cv[i].W, cv[i].b, cv[i].g, cv[i].bb, cv[i].m, cv[i].v,
          xb2, BSZ, L);
      int L2 = (L - 1) / 2 + 1;   // 200->100->50
      maxpool_kernel<<<cdiv(BSZ * L2 * DM, 256), 256, 0, stream>>>(
          xb2, xe, BSZ, L, L2);
      L = L2;
    }
  }
  add_ln(xe, nullptr, enc_ng, enc_nb, crossb, BSZ * L);
  const int Lc = L;  // 50

  // ================= Decoder =================
  const int Ld = 200;
  embed_kernel<<<cdiv(BSZ * Ld * DM, 256), 256, 0, stream>>>(
      x_dec, dec_emb_W, y1, BSZ, Ld, 12);
  for (int i = 0; i < 2; ++i) {
    int M = BSZ * Ld;
    // causal ProbSparse self-attention (mix=True)
    gemm(y1, dec[i].sa.Wq, dec[i].sa.bq, qb, M, DM, DM, 0);
    gemm(y1, dec[i].sa.Wk, dec[i].sa.bk, kb, M, DM, DM, 0);
    gemm(y1, dec[i].sa.Wv, dec[i].sa.bv, vb, M, DM, DM, 0);
    prob_attn(qb, kb, vb, ctx, Ld, /*causal=*/1, (unsigned)(100 + i));
    mix_kernel<<<cdiv(M * DM, 256), 256, 0, stream>>>(ctx, xb2, BSZ, Ld);
    gemm(xb2, dec[i].sa.Wo, dec[i].sa.bo, pout, M, DM, DM, 0);
    add_ln(y1, pout, dec[i].l1g, dec[i].l1b, y2, M);
    // dense cross-attention
    gemm(y2,     dec[i].ca.Wq, dec[i].ca.bq, qb, M,        DM, DM, 0);
    gemm(crossb, dec[i].ca.Wk, dec[i].ca.bk, kb, BSZ * Lc, DM, DM, 0);
    gemm(crossb, dec[i].ca.Wv, dec[i].ca.bv, vb, BSZ * Lc, DM, DM, 0);
    full_attn_kernel<<<BSZ * NH * Ld, 32, 0, stream>>>(
        qb, kb, vb, ctx, BSZ, Ld, Lc);
    gemm(ctx, dec[i].ca.Wo, dec[i].ca.bo, pout, M, DM, DM, 0);
    add_ln(y2, pout, dec[i].l2g, dec[i].l2b, y1, M);
    // FFN
    gemm(y1, dec[i].W1, dec[i].b1, hid, M, 512, DM, /*gelu=*/1);
    gemm(hid, dec[i].W2, dec[i].b2, pout, M, DM, 512, 0);
    add_ln(y1, pout, dec[i].l3g, dec[i].l3b, y1, M);  // safe: row-local RAW only
  }
  add_ln(y1, nullptr, dec_ng, dec_nb, y2, BSZ * Ld);
  proj_kernel<<<cdiv(BSZ * Ld * 12, 256), 256, 0, stream>>>(
      y2, projW, projb, (float*)d_out, BSZ, Ld);
}
